// GistExtractor_7834020348583
// MI455X (gfx1250) — matrix operand order repaired and checked by reading, not compiled, not run on hardware
//
#include <hip/hip_runtime.h>
#include <hip/hip_bf16.h>
#include <math.h>

// Problem constants (match reference)
#define BB 4
#define TT 2048
#define DD 1024
#define HH 16
#define DH 64
#define T2 1024
#define NC 32            // scan chunks per (b,h)
#define CH (TT / NC)     // 64 timesteps per chunk

typedef __attribute__((ext_vector_type(16))) __bf16 v16bf;
typedef __attribute__((ext_vector_type(8)))  float  v8f;
typedef int v4i_vec __attribute__((__vector_size__(16)));   // builtin's b128 unit

// CDNA5 async global->LDS DMA (ASYNCcnt-tracked), if the toolchain exposes it.
#if defined(__HIP_DEVICE_COMPILE__) && __has_builtin(__builtin_amdgcn_global_load_async_to_lds_b128)
#define USE_ASYNC_LDS 1
#define AS1P(p) ((__attribute__((address_space(1))) v4i_vec*)(unsigned long long)(p))
#define AS3P(p) ((__attribute__((address_space(3))) v4i_vec*)(unsigned long long)(p))
#if __has_builtin(__builtin_amdgcn_s_wait_asynccnt)
#define WAIT_ASYNC() __builtin_amdgcn_s_wait_asynccnt(0)
#else
#define WAIT_ASYNC() asm volatile("s_wait_asynccnt 0x0" ::: "memory")
#endif
#else
#define USE_ASYNC_LDS 0
#endif

// ---------------------------------------------------------------------------
// Pack A (row-major f32 MxK) -> WMMA bf16 A-fragment layout [kt][mt][lane][e]
// A 16x32 bf16 fragment: lane holds row m=lane&15; element e maps to
// K = (e&7) + (e<8?0:16) + (lane>=16?8:0)   (CDNA5 ISA 7.12.2)
// ---------------------------------------------------------------------------
__global__ void pack_a_bf16(const float* __restrict__ A, __bf16* __restrict__ Ap,
                            int M, int K) {
    int tid = blockIdx.x * blockDim.x + threadIdx.x;
    int lane = tid & 31;
    int grp  = tid >> 5;
    int MT = M >> 4;
    int mt = grp % MT;
    int kt = grp / MT;
    if (kt >= (K >> 5)) return;
    int r = mt * 16 + (lane & 15);
    int khalf = (lane >> 4) * 8;
    const float* s0 = A + (size_t)r * K + kt * 32 + khalf;
    const float* s1 = s0 + 16;
    __bf16 tmp[16];
#pragma unroll
    for (int e = 0; e < 8; ++e)  tmp[e]     = (__bf16)s0[e];
#pragma unroll
    for (int e = 0; e < 8; ++e)  tmp[8 + e] = (__bf16)s1[e];
    uint4* dst = (uint4*)(Ap + ((size_t)(kt * MT + mt) * 32 + lane) * 16);
    dst[0] = ((uint4*)tmp)[0];
    dst[1] = ((uint4*)tmp)[1];
}

// ---------------------------------------------------------------------------
// Pack B (row-major f32 KxN) -> WMMA bf16 B-fragment layout [kt][nt][lane][e]
// B 32x16 fragment: lane holds column n=lane&15; K = (e&7) + (e<8?0:16) + (lane>=16?8:0)
// ---------------------------------------------------------------------------
__global__ void pack_b_bf16(const float* __restrict__ Bsrc, __bf16* __restrict__ Bp,
                            int K, int N) {
    int tid = blockIdx.x * blockDim.x + threadIdx.x;
    int lane = tid & 31;
    int grp  = tid >> 5;
    int NT = N >> 4;
    int nt = grp % NT;
    int kt = grp / NT;
    if (kt >= (K >> 5)) return;
    int col = nt * 16 + (lane & 15);
    int k0  = kt * 32 + (lane >> 4) * 8;
    __bf16 tmp[16];
#pragma unroll
    for (int e = 0; e < 16; ++e)
        tmp[e] = (__bf16)Bsrc[(size_t)(k0 + e + (e >= 8 ? 8 : 0)) * N + col];
    uint4* dst = (uint4*)(Bp + ((size_t)(kt * NT + nt) * 32 + lane) * 16);
    dst[0] = ((uint4*)tmp)[0];
    dst[1] = ((uint4*)tmp)[1];
}

// ---------------------------------------------------------------------------
// wq[h][d] = (1/sqrt(DH)) * sum_e Wk[d][h*DH+e] * q[h*DH+e]
// ---------------------------------------------------------------------------
__global__ void wq_fold_kernel(const float* __restrict__ Wk,
                               const float* __restrict__ q,
                               float* __restrict__ wq) {
    int tid = blockIdx.x * blockDim.x + threadIdx.x;   // D*H threads
    if (tid >= DD * HH) return;
    int d = tid & (DD - 1);
    int h = tid >> 10;
    const float4* wr = (const float4*)(Wk + (size_t)d * DD + h * DH);
    const float4* qv = (const float4*)(q + h * DH);
    float acc = 0.f;
#pragma unroll
    for (int i = 0; i < DH / 4; ++i) {
        float4 a = wr[i], b = qv[i];
        acc += a.x * b.x + a.y * b.y + a.z * b.z + a.w * b.w;
    }
    wq[(size_t)h * DD + d] = acc * 0.125f;   // 1/sqrt(64)
}

// ---------------------------------------------------------------------------
// scores[b][h][t] = x[b,t,:] . wq[h,:]   (x rows staged in LDS, 16 rows/block)
// ---------------------------------------------------------------------------
__global__ void scores_kernel(const float* __restrict__ x,
                              const float* __restrict__ wq,
                              float* __restrict__ scores) {
    __shared__ float xs[16 * DD];   // 64 KB of the 320 KB WGP LDS
    int tid = threadIdx.x;          // 256
    size_t row0 = (size_t)blockIdx.x * 16;
    const float4* src4 = (const float4*)(x + row0 * DD);
    float4* xs4 = (float4*)xs;
#pragma unroll
    for (int i = 0; i < 16; ++i)
        xs4[tid + i * 256] = src4[tid + i * 256];
    __syncthreads();
    int tl = tid >> 4;              // local row 0..15
    int h  = tid & 15;              // head
    const float4* xr = (const float4*)(xs + tl * DD);
    const float4* wr = (const float4*)(wq + (size_t)h * DD);
    float acc = 0.f;
#pragma unroll 8
    for (int i = 0; i < DD / 4; ++i) {
        float4 a = xr[i], b = wr[i];
        acc += a.x * b.x + a.y * b.y + a.z * b.z + a.w * b.w;
    }
    size_t m = row0 + tl;           // global (b*T + t)
    int b = (int)(m >> 11);
    int t = (int)(m & (TT - 1));
    scores[((size_t)(b * HH + h)) * TT + t] = acc;
}

// ---------------------------------------------------------------------------
// Scan phase A: per (b,h,chunk) softmax-state totals (m, Z, A[64]).
// ---------------------------------------------------------------------------
__global__ void scan_phaseA(const float* __restrict__ scores,
                            const float* __restrict__ v,
                            float* __restrict__ cs) {
    __shared__ float ss[CH];
    int c  = blockIdx.x % NC;
    int bh = blockIdx.x / NC;
    int b  = bh >> 4, h = bh & 15;
    int dim = threadIdx.x;          // 0..63
    const float* srow = scores + (size_t)bh * TT + c * CH;
    ss[dim] = srow[dim];
    __syncthreads();
    float m = -3.0e38f, Z = 0.f, A = 0.f;
    const float* vb = v + ((size_t)b * TT + c * CH) * DD + h * DH + dim;
    for (int tl = 0; tl < CH; ++tl) {
        float s = ss[tl];
        if (s > m) { float f = __expf(m - s); Z *= f; A *= f; m = s; }
        float w = __expf(s - m);
        Z += w;
        A += w * vb[(size_t)tl * DD];
    }
    float* out = cs + (size_t)(bh * NC + c) * 66;
    if (dim == 0) { out[0] = m; out[1] = Z; }
    out[2 + dim] = A;
}

// ---------------------------------------------------------------------------
// Scan phase B: exclusive prefix combine of chunk states per (b,h).
// ---------------------------------------------------------------------------
__global__ void scan_phaseB(const float* __restrict__ cs,
                            float* __restrict__ carry) {
    int bh = blockIdx.x;
    int dim = threadIdx.x;          // 0..63
    float cm = -3.0e38f, cZ = 0.f, cA = 0.f;
    for (int c = 0; c < NC; ++c) {
        float* out = carry + (size_t)(bh * NC + c) * 66;
        if (dim == 0) { out[0] = cm; out[1] = cZ; }
        out[2 + dim] = cA;
        const float* in = cs + (size_t)(bh * NC + c) * 66;
        float mc = in[0], Zc = in[1], Ac = in[2 + dim];
        float nm = fmaxf(cm, mc);
        float f0 = __expf(cm - nm), f1 = __expf(mc - nm);
        cZ = cZ * f0 + Zc * f1;
        cA = cA * f0 + Ac * f1;
        cm = nm;
    }
}

// ---------------------------------------------------------------------------
// Scan phase C: seeded local scan; emit pooled[b,j,h,:] = A/Z at t = 2j.
// ---------------------------------------------------------------------------
__global__ void scan_phaseC(const float* __restrict__ scores,
                            const float* __restrict__ v,
                            const float* __restrict__ carry,
                            float* __restrict__ pooled) {
    __shared__ float ss[CH];
    int c  = blockIdx.x % NC;
    int bh = blockIdx.x / NC;
    int b  = bh >> 4, h = bh & 15;
    int dim = threadIdx.x;
    const float* srow = scores + (size_t)bh * TT + c * CH;
    ss[dim] = srow[dim];
    __syncthreads();
    const float* cin = carry + (size_t)(bh * NC + c) * 66;
    float m = cin[0], Z = cin[1], A = cin[2 + dim];
    const float* vb = v + ((size_t)b * TT + c * CH) * DD + h * DH + dim;
    float* pb = pooled + ((size_t)b * T2 + c * (CH / 2)) * DD + h * DH + dim;
    for (int tl = 0; tl < CH; ++tl) {
        float s = ss[tl];
        if (s > m) { float f = __expf(m - s); Z *= f; A *= f; m = s; }
        float w = __expf(s - m);
        Z += w;
        A += w * vb[(size_t)tl * DD];
        if ((tl & 1) == 0)                     // t = 2j inclusive
            pb[(size_t)(tl >> 1) * DD] = A / Z;
    }
}

// ---------------------------------------------------------------------------
// LayerNorm (in-place over pooled rows) + mag head: sigmoid(pooled_n.Wm + bm)
// ---------------------------------------------------------------------------
__global__ void ln_mag_kernel(float* __restrict__ pooled,
                              const float* __restrict__ g,
                              const float* __restrict__ bb,
                              const float* __restrict__ Wm,
                              const float* __restrict__ bm,
                              float* __restrict__ mag) {
    __shared__ float red[256];
    int tid = threadIdx.x;
    float4* p4 = (float4*)(pooled + (size_t)blockIdx.x * DD);
    float4 xv = p4[tid];
    red[tid] = xv.x + xv.y + xv.z + xv.w; __syncthreads();
    for (int s = 128; s > 0; s >>= 1) { if (tid < s) red[tid] += red[tid + s]; __syncthreads(); }
    float mu = red[0] * (1.f / DD);
    __syncthreads();
    float dx = xv.x - mu, dy = xv.y - mu, dz = xv.z - mu, dw = xv.w - mu;
    red[tid] = dx * dx + dy * dy + dz * dz + dw * dw; __syncthreads();
    for (int s = 128; s > 0; s >>= 1) { if (tid < s) red[tid] += red[tid + s]; __syncthreads(); }
    float rstd = rsqrtf(red[0] * (1.f / DD) + 1e-5f);
    __syncthreads();
    float4 gv = ((const float4*)g)[tid];
    float4 bv = ((const float4*)bb)[tid];
    float4 yv;
    yv.x = dx * rstd * gv.x + bv.x;
    yv.y = dy * rstd * gv.y + bv.y;
    yv.z = dz * rstd * gv.z + bv.z;
    yv.w = dw * rstd * gv.w + bv.w;
    p4[tid] = yv;
    float4 wv = ((const float4*)Wm)[tid];
    red[tid] = yv.x * wv.x + yv.y * wv.y + yv.z * wv.z + yv.w * wv.w;
    __syncthreads();
    for (int s = 128; s > 0; s >>= 1) { if (tid < s) red[tid] += red[tid + s]; __syncthreads(); }
    if (tid == 0) {
        float z = red[0] + bm[0];
        mag[blockIdx.x] = 1.f / (1.f + __expf(-z));
    }
}

// ---------------------------------------------------------------------------
// WMMA bf16 GEMM: C(MxN,f32) = Apack x Bpack (+ optional column bias)
// Block 256 thr (8 waves), 128x256x32 tile, double-buffered LDS (48 KB).
// Wave grid 2x4; each wave owns 64x64 = 4x4 tiles -> 16 WMMA per K-step.
// __launch_bounds__(256, 1): CDNA5 waves can hold up to 1024 VGPRs; let the
// 128 accumulator + 64 fragment VGPRs live in registers (round-3 asm showed
// the default 128-VGPR cap spilling accumulators to scratch every K-step).
// ---------------------------------------------------------------------------
__global__ void __launch_bounds__(256, 1)
gemm_bf16_wmma(const __bf16* __restrict__ Ap,
               const __bf16* __restrict__ Bp,
               float* __restrict__ C,
               int M, int N, int K,
               const float* __restrict__ bias) {
    __shared__ __attribute__((aligned(32))) __bf16 As[2][8 * 512];    // 8 KB/buf
    __shared__ __attribute__((aligned(32))) __bf16 Bs[2][16 * 512];   // 16 KB/buf
    const int tid  = threadIdx.x;
    const int lane = tid & 31;
    const int w    = tid >> 5;
    const int wm   = w >> 2;      // 0..1
    const int wn   = w & 3;       // 0..3
    const int MT = M >> 4, NT = N >> 4, KT = K >> 5;
    const int bm0 = blockIdx.y * 8;    // first m-tile of block
    const int bn0 = blockIdx.x * 16;   // first n-tile of block

    v8f acc[4][4];
#pragma unroll
    for (int i = 0; i < 4; ++i)
#pragma unroll
        for (int j = 0; j < 4; ++j) {
            v8f z = {0.f, 0.f, 0.f, 0.f, 0.f, 0.f, 0.f, 0.f};
            acc[i][j] = z;
        }

    auto stage = [&](int nb, int kt) {
        const __bf16* gA = Ap + ((size_t)kt * MT + bm0) * 512;   //  8 KB contiguous
        const __bf16* gB = Bp + ((size_t)kt * NT + bn0) * 512;   // 16 KB contiguous
#if USE_ASYNC_LDS
#pragma unroll
        for (int i = 0; i < 2; ++i)
            __builtin_amdgcn_global_load_async_to_lds_b128(
                AS1P(gA + (size_t)(tid + i * 256) * 8),
                AS3P(&As[nb][(tid + i * 256) * 8]), 0, 0);
#pragma unroll
        for (int i = 0; i < 4; ++i)
            __builtin_amdgcn_global_load_async_to_lds_b128(
                AS1P(gB + (size_t)(tid + i * 256) * 8),
                AS3P(&Bs[nb][(tid + i * 256) * 8]), 0, 0);
#else
        const uint4* sA = (const uint4*)gA;
        const uint4* sB = (const uint4*)gB;
        uint4 ra[2], rb[4];
#pragma unroll
        for (int i = 0; i < 2; ++i) ra[i] = sA[tid + i * 256];
#pragma unroll
        for (int i = 0; i < 4; ++i) rb[i] = sB[tid + i * 256];
        uint4* dA = (uint4*)(As[nb]);
        uint4* dB = (uint4*)(Bs[nb]);
#pragma unroll
        for (int i = 0; i < 2; ++i) dA[tid + i * 256] = ra[i];
#pragma unroll
        for (int i = 0; i < 4; ++i) dB[tid + i * 256] = rb[i];
#endif
    };

    stage(0, 0);
    int buf = 0;
    for (int kt = 0; kt < KT; ++kt) {
#if USE_ASYNC_LDS
        WAIT_ASYNC();        // drain this wave's async LDS writes
#endif
        __syncthreads();
        if (kt + 1 < KT) stage(buf ^ 1, kt + 1);
        v16bf af[4];
#pragma unroll
        for (int i = 0; i < 4; ++i)
            af[i] = *(const v16bf*)&As[buf][((wm * 4 + i) * 32 + lane) * 16];
        v16bf bfr[4];
#pragma unroll
        for (int j = 0; j < 4; ++j)
            bfr[j] = *(const v16bf*)&Bs[buf][((wn * 4 + j) * 32 + lane) * 16];
#pragma unroll
        for (int i = 0; i < 4; ++i)
#pragma unroll
            for (int j = 0; j < 4; ++j)
                acc[i][j] = __builtin_amdgcn_wmma_f32_16x16x32_bf16(
                    false, af[i], false, bfr[j], (short)0, acc[i][j], false, false);
        buf ^= 1;
    }

    // epilogue: C/D layout — VGPR g: rows g (lanes 0-15) / g+8 (lanes 16-31)
    const int half = lane >> 4;
    const int nlan = lane & 15;
#pragma unroll
    for (int i = 0; i < 4; ++i) {
#pragma unroll
        for (int j = 0; j < 4; ++j) {
            int r0 = (bm0 + wm * 4 + i) * 16 + half * 8;
            int c0 = (bn0 + wn * 4 + j) * 16 + nlan;
            float bvv = bias ? bias[c0] : 0.f;
#pragma unroll
            for (int gg = 0; gg < 8; ++gg)
                C[(size_t)(r0 + gg) * N + c0] = acc[i][j][gg] + bvv;
        }
    }
}

// ---------------------------------------------------------------------------
extern "C" void kernel_launch(void* const* d_in, const int* in_sizes, int n_in,
                              void* d_out, int out_size, void* d_ws, size_t ws_size,
                              hipStream_t stream) {
    const float* x     = (const float*)d_in[0];
    const float* query = (const float*)d_in[1];
    const float* Wk    = (const float*)d_in[2];
    const float* Wv    = (const float*)d_in[3];
    const float* Wt    = (const float*)d_in[4];
    const float* bt    = (const float*)d_in[5];
    const float* Wm    = (const float*)d_in[6];
    const float* bm    = (const float*)d_in[7];
    const float* ln_g  = (const float*)d_in[8];
    const float* ln_b  = (const float*)d_in[9];

    float* theta = (float*)d_out;                       // (B,T2,D/2)
    float* mag   = theta + (size_t)BB * T2 * (DD / 2);  // (B,T2,1)

    const int M1 = BB * TT;   // 8192
    const int M2 = BB * T2;   // 4096
    const int N2 = DD / 2;    // 512

    char* ws = (char*)d_ws;
    size_t off = 0;
    auto carve = [&](size_t bytes) -> void* {
        void* p = ws + off;
        off += (bytes + 255) & ~(size_t)255;
        return p;
    };
    __bf16* ApX = (__bf16*)carve((size_t)M1 * DD * sizeof(__bf16));        // 16.8 MB
    __bf16* BpV = (__bf16*)carve((size_t)DD * DD * sizeof(__bf16));        //  2.1 MB
    __bf16* BpT = (__bf16*)carve((size_t)DD * N2 * sizeof(__bf16));        //  1.0 MB
    float*  wq  = (float*)carve((size_t)HH * DD * sizeof(float));          // 64 KB
    float*  sc  = (float*)carve((size_t)BB * HH * TT * sizeof(float));     // 512 KB
    float*  v   = (float*)carve((size_t)M1 * DD * sizeof(float));          // 33.6 MB
    float*  cs  = (float*)carve((size_t)BB * HH * NC * 66 * sizeof(float));
    float*  car = (float*)carve((size_t)BB * HH * NC * 66 * sizeof(float));
    float*  pooled = (float*)carve((size_t)M2 * DD * sizeof(float));       // 16.8 MB
    __bf16* ApP = (__bf16*)carve((size_t)M2 * DD * sizeof(__bf16));        //  8.4 MB

    // 1) pack activations & weights to WMMA fragment layouts
    {
        int thr = (DD / 32) * (M1 / 16) * 32;
        hipLaunchKernelGGL(pack_a_bf16, dim3(thr / 256), dim3(256), 0, stream, x, ApX, M1, DD);
    }
    {
        int thr = (DD / 32) * (DD / 16) * 32;
        hipLaunchKernelGGL(pack_b_bf16, dim3(thr / 256), dim3(256), 0, stream, Wv, BpV, DD, DD);
    }
    {
        int thr = (DD / 32) * (N2 / 16) * 32;
        hipLaunchKernelGGL(pack_b_bf16, dim3(thr / 256), dim3(256), 0, stream, Wt, BpT, DD, N2);
    }
    // 2) fold Wk,q -> wq; scores = x @ wq  (k never materialized)
    hipLaunchKernelGGL(wq_fold_kernel, dim3((DD * HH) / 256), dim3(256), 0, stream, Wk, query, wq);
    hipLaunchKernelGGL(scores_kernel, dim3(M1 / 16), dim3(256), 0, stream, x, wq, sc);
    // 3) v = x @ Wv  (WMMA bf16), 128x256 block tiles
    hipLaunchKernelGGL(gemm_bf16_wmma, dim3(DD / 256, M1 / 128), dim3(256), 0, stream,
                       ApX, BpV, v, M1, DD, DD, (const float*)nullptr);
    // 4) chunked softmax prefix scan -> pooled
    hipLaunchKernelGGL(scan_phaseA, dim3(BB * HH * NC), dim3(64), 0, stream, sc, v, cs);
    hipLaunchKernelGGL(scan_phaseB, dim3(BB * HH), dim3(64), 0, stream, cs, car);
    hipLaunchKernelGGL(scan_phaseC, dim3(BB * HH * NC), dim3(64), 0, stream, sc, v, car, pooled);
    // 5) LayerNorm (in-place) + mag head
    hipLaunchKernelGGL(ln_mag_kernel, dim3(M2), dim3(256), 0, stream,
                       pooled, ln_g, ln_b, Wm, bm, mag);
    // 6) theta = pooled_n @ Wt + bt  (WMMA bf16)
    {
        int thr = (DD / 32) * (M2 / 16) * 32;
        hipLaunchKernelGGL(pack_a_bf16, dim3(thr / 256), dim3(256), 0, stream, pooled, ApP, M2, DD);
    }
    hipLaunchKernelGGL(gemm_bf16_wmma, dim3(N2 / 256, M2 / 128), dim3(256), 0, stream,
                       ApP, BpT, theta, M2, N2, DD, bt);
}
